// LocalGraph_4355096839073
// MI455X (gfx1250) — matrix operand order repaired and checked by reading, not compiled, not run on hardware
//
#include <hip/hip_runtime.h>
#include <hip/hip_bf16.h>
#include <math.h>

// ---------------------------------------------------------------------------
// LocalGraph pipeline for MI455X (gfx1250, wave32)
//   scores = log(sigmoid(<l2n(sub), l2n(emb)>)) + gumbel ;  seeds = top-4096
//   sub    = (spmm(spmm(E)-E) - deg*E) / (spmm(deg) - deg + eps)
// SpMM: COO + f32 L2 atomics (entire table is L2-resident: 25.6MB << 192MB).
// Scoring: per-wave 16-row tile, 2x v_wmma_f32_16x16x32_bf16, diagonal of C.
// Top-k: 3-level radix select + single-block bitonic sort.
// ---------------------------------------------------------------------------

typedef __attribute__((ext_vector_type(16))) __bf16 v16bf;
typedef __attribute__((ext_vector_type(8)))  float  v8f;

#define SEEDK 4096
#define DEPS  1e-8f

__device__ __forceinline__ unsigned keyOf(float f) {
  unsigned u = __float_as_uint(f);
  return (u & 0x80000000u) ? ~u : (u | 0x80000000u);   // monotone: bigger float -> bigger key
}
__device__ __forceinline__ unsigned pcg_hash(unsigned v) {
  v = v * 747796405u + 2891336453u;
  unsigned w = ((v >> ((v >> 28u) + 4u)) ^ v) * 277803737u;
  return (w >> 22u) ^ w;
}

// ---- workspace init: zero accumulators/histograms, set sel[1]=SEEDK --------
__global__ void init_ws_kernel(unsigned* ws, long long zeroCount, long long selRemIdx, unsigned kval) {
  long long stride = (long long)gridDim.x * blockDim.x;
  for (long long i = (long long)blockIdx.x * blockDim.x + threadIdx.x; i < zeroCount; i += stride)
    ws[i] = (i == selRemIdx) ? kval : 0u;
}

// ---- degree + scalar SpMM --------------------------------------------------
__global__ void degree_kernel(const int* __restrict__ rows, float* __restrict__ order, int E) {
  int e = blockIdx.x * blockDim.x + threadIdx.x;
  if (e < E) atomicAdd(&order[rows[e]], 1.0f);
}
__global__ void spmm_ord_kernel(const int* __restrict__ rows, const int* __restrict__ cols,
                                const float* __restrict__ order, float* __restrict__ ordAcc, int E) {
  int e = blockIdx.x * blockDim.x + threadIdx.x;
  if (e < E) atomicAdd(&ordAcc[rows[e]], order[cols[e]]);
}

// ---- feature SpMM: 16 threads/edge, float4 gather + 4 f32 atomics ----------
__global__ void spmm_feat_kernel(const int* __restrict__ rows, const int* __restrict__ cols,
                                 const float* __restrict__ x, float* __restrict__ acc, int E) {
  long long tid = (long long)blockIdx.x * blockDim.x + threadIdx.x;
  if (tid >= (long long)E * 16) return;
  int e  = (int)(tid >> 4);
  int c4 = ((int)tid & 15) * 4;
  const float4 v = *(const float4*)(x + (long long)cols[e] * 64 + c4);
  float* dst = acc + (long long)rows[e] * 64 + c4;
  atomicAdd(dst + 0, v.x); atomicAdd(dst + 1, v.y);
  atomicAdd(dst + 2, v.z); atomicAdd(dst + 3, v.w);
}

// ---- fst = spmm(E) - E ; also re-zero bufA for reuse as acc2 ---------------
__global__ void fst_kernel(float* __restrict__ bufA, const float* __restrict__ embeds,
                           float* __restrict__ bufB, long long n) {
  long long i = (long long)blockIdx.x * blockDim.x + threadIdx.x;
  if (i < n) { float t = bufA[i]; bufB[i] = t - embeds[i]; bufA[i] = 0.0f; }
}

// ---- WMMA scoring: one wave per 16-node tile -------------------------------
__global__ __launch_bounds__(256)
void score_wmma_kernel(const float* __restrict__ acc2, const float* __restrict__ order,
                       const float* __restrict__ ordAcc, const float* __restrict__ embeds,
                       float* __restrict__ scores, int N, int ntiles) {
  __shared__ __bf16 ldsA[8][16][64];   // normalized subgraph rows  (bf16)
  __shared__ __bf16 ldsB[8][16][64];   // normalized embedding rows (bf16)
  const int lane = threadIdx.x & 31;
  const int wave = threadIdx.x >> 5;
  int tile = blockIdx.x * 8 + wave;
  const bool active = tile < ntiles;
  if (!active) tile = ntiles - 1;      // no early return: keep EXEC uniform for WMMA
  const int base = tile * 16;

  // Phase A: 2 lanes per node, 32 dims each; two-pass (sumsq, then normalize)
  {
    const int node = lane >> 1;
    const int h    = lane & 1;
    int g  = base + node;
    int gc = (g < N) ? g : (N - 1);
    float ord    = order[gc];
    float invDen = 1.0f / (ordAcc[gc] - ord + DEPS);
    const float* aRow = acc2   + (long long)gc * 64 + h * 32;
    const float* eRow = embeds + (long long)gc * 64 + h * 32;
    float ss_s = 0.f, ss_e = 0.f;
    #pragma unroll
    for (int d = 0; d < 32; ++d) {
      float e = eRow[d];
      float s = (aRow[d] - ord * e) * invDen;
      ss_s += s * s; ss_e += e * e;
    }
    ss_s += __shfl_xor(ss_s, 1, 32);
    ss_e += __shfl_xor(ss_e, 1, 32);
    float scS = 1.0f / fmaxf(sqrtf(ss_s), 1e-12f);
    float scE = 1.0f / fmaxf(sqrtf(ss_e), 1e-12f);
    #pragma unroll
    for (int d = 0; d < 32; ++d) {
      float e = eRow[d];
      float s = (aRow[d] - ord * e) * invDen;
      ldsA[wave][node][h * 32 + d] = (__bf16)(s * scS);
      ldsB[wave][node][h * 32 + d] = (__bf16)(e * scE);
    }
  }
  __syncthreads();

  // Phase B: build 16x32 bf16 A and 32x16 bf16 B fragments (contiguous k = 16h + s, same order for A and B)
  const int row = lane & 15;
  const int kbA = (lane < 16) ? 0 : 16;   // A: lanes0-15 K=0..15, lanes16-31 K=16..31
  const int kbB = (lane < 16) ? 0 : 16;   // B: lanes0-15 K=0..15, lanes16-31 K=16..31 (N = lane&15)
  v8f c = {};
  #pragma unroll
  for (int chunk = 0; chunk < 64; chunk += 32) {
    v16bf a, b;
    #pragma unroll
    for (int t = 0; t < 16; ++t) {
      a[t] = ldsA[wave][row][chunk + kbA + t];
      b[t] = ldsB[wave][row][chunk + kbB + t];
    }
    c = __builtin_amdgcn_wmma_f32_16x16x32_bf16(false, a, false, b, (short)0, c, false, false);
  }

  // Diagonal of C: i<8 -> lane i, vgpr i ; i>=8 -> lane i+16, vgpr i-8
  const bool isLo = (lane < 8);
  const bool isHi = (lane >= 24);
  int r = isLo ? lane : (lane - 24);
  float dot = c[0];
  #pragma unroll
  for (int q = 1; q < 8; ++q) if (r == q) dot = c[q];

  if (active && (isLo || isHi)) {
    int nd  = isLo ? lane : (lane - 16);
    int gid = base + nd;
    if (gid < N) {
      float sgm   = 1.0f / (1.0f + __expf(-dot));
      unsigned h  = pcg_hash((unsigned)gid + 42u * 0x9E3779B9u);
      float u     = fmaxf((float)(h >> 8) * (1.0f / 16777216.0f), 1e-8f);
      float noise = -__logf(-__logf(u));
      scores[gid] = __logf(sgm) + noise;
    }
  }
}

// ---- top-k: 3-level radix select ------------------------------------------
__global__ void hist_kernel(const float* __restrict__ scores, unsigned* __restrict__ hist,
                            const unsigned* __restrict__ sel, int level, int n) {
  int stride = gridDim.x * blockDim.x;
  unsigned prefix = sel[0];
  for (int i = blockIdx.x * blockDim.x + threadIdx.x; i < n; i += stride) {
    unsigned k = keyOf(scores[i]);
    if (level == 0)      atomicAdd(&hist[k >> 20], 1u);
    else if (level == 1) { if ((k >> 20) == (prefix >> 20)) atomicAdd(&hist[(k >> 8) & 0xFFFu], 1u); }
    else                 { if ((k >> 8)  == (prefix >> 8))  atomicAdd(&hist[k & 0xFFu], 1u); }
  }
}

__global__ void scan_kernel(unsigned* __restrict__ hist, unsigned* __restrict__ sel, int level) {
  int nb = (level == 2) ? 256 : 4096;
  if (threadIdx.x == 0) {
    unsigned K = sel[1];
    unsigned acc = 0; int T = 0;
    for (int b = nb - 1; b >= 0; --b) { acc += hist[b]; if (acc >= K) { T = b; break; } }
    unsigned cg = acc - hist[T];                 // strictly-greater buckets
    if (level == 0)      sel[0]  = ((unsigned)T) << 20;
    else if (level == 1) sel[0] |= ((unsigned)T) << 8;
    else                 sel[0] |= (unsigned)T;
    unsigned rem = K - cg;
    sel[1] = rem;
    if (level == 2) { sel[5] = rem; sel[4] = (unsigned)SEEDK - rem; }
  }
  __syncthreads();
  for (int b = threadIdx.x; b < nb; b += blockDim.x) hist[b] = 0;  // clear for next level
}

__global__ void compact_kernel(const float* __restrict__ scores, unsigned* __restrict__ sel,
                               unsigned* __restrict__ candKey, int* __restrict__ candIdx, int n) {
  int i = blockIdx.x * blockDim.x + threadIdx.x;
  if (i >= n) return;
  unsigned k = keyOf(scores[i]);
  unsigned thr = sel[0];
  if (k > thr) {
    unsigned p = atomicAdd(&sel[2], 1u);
    if (p < (unsigned)SEEDK) { candKey[p] = k; candIdx[p] = i; }
  } else if (k == thr) {
    unsigned t = atomicAdd(&sel[3], 1u);
    if (t < sel[5]) { unsigned p = sel[4] + t; candKey[p] = k; candIdx[p] = i; }
  }
}

// ---- single-block bitonic sort (descending by key, idx-ascending ties) -----
__global__ __launch_bounds__(1024)
void sort_kernel(const unsigned* __restrict__ candKey, const int* __restrict__ candIdx,
                 float* __restrict__ outSeeds) {
  __shared__ unsigned sk[SEEDK];
  __shared__ int      si[SEEDK];
  for (int t = threadIdx.x; t < SEEDK; t += blockDim.x) { sk[t] = candKey[t]; si[t] = candIdx[t]; }
  __syncthreads();
  for (int k = 2; k <= SEEDK; k <<= 1) {
    for (int j = k >> 1; j > 0; j >>= 1) {
      for (int t = threadIdx.x; t < SEEDK; t += blockDim.x) {
        int ixj = t ^ j;
        if (ixj > t) {
          bool desc = ((t & k) == 0);
          unsigned a = sk[t], b = sk[ixj];
          bool lt = (a < b) || (a == b && si[t] > si[ixj]);
          if (lt == desc) {
            sk[t] = b; sk[ixj] = a;
            int tmp = si[t]; si[t] = si[ixj]; si[ixj] = tmp;
          }
        }
      }
      __syncthreads();
    }
  }
  for (int t = threadIdx.x; t < SEEDK; t += blockDim.x) outSeeds[t] = (float)si[t];
}

// ---------------------------------------------------------------------------
extern "C" void kernel_launch(void* const* d_in, const int* in_sizes, int n_in,
                              void* d_out, int out_size, void* d_ws, size_t ws_size,
                              hipStream_t stream) {
  const int*   rows   = (const int*)d_in[0];
  const int*   cols   = (const int*)d_in[1];
  const float* embeds = (const float*)d_in[2];
  const int E = in_sizes[0];
  const int N = in_sizes[2] / 64;
  const long long ND = (long long)N * 64;

  // workspace layout (32-bit words); bufB at the end is fully overwritten -> not zeroed
  const long long o_order   = 0;
  const long long o_ordAcc  = o_order   + N;
  const long long o_bufA    = o_ordAcc  + N;
  const long long o_hist    = o_bufA    + ND;
  const long long o_sel     = o_hist    + 4096;
  const long long o_candK   = o_sel     + 16;
  const long long o_candI   = o_candK   + SEEDK;
  const long long o_bufB    = o_candI   + SEEDK;

  float*    wsf     = (float*)d_ws;
  unsigned* wsu     = (unsigned*)d_ws;
  float*    order   = wsf + o_order;
  float*    ordAcc  = wsf + o_ordAcc;
  float*    bufA    = wsf + o_bufA;
  float*    bufB    = wsf + o_bufB;
  unsigned* hist    = wsu + o_hist;
  unsigned* sel     = wsu + o_sel;
  unsigned* candKey = wsu + o_candK;
  int*      candIdx = (int*)(wsu + o_candI);

  float* scores   = (float*)d_out;       // [N]
  float* outSeeds = (float*)d_out + N;   // [SEEDK]

  // 1) init
  init_ws_kernel<<<4096, 256, 0, stream>>>(wsu, o_bufB, o_sel + 1, (unsigned)SEEDK);
  // 2) degree
  degree_kernel<<<(E + 255) / 256, 256, 0, stream>>>(rows, order, E);
  // 3) acc1 = spmm(embeds)
  {
    long long tt = (long long)E * 16;
    spmm_feat_kernel<<<(unsigned)((tt + 255) / 256), 256, 0, stream>>>(rows, cols, embeds, bufA, E);
  }
  // 4) fst = acc1 - embeds ; acc1 <- 0
  fst_kernel<<<(unsigned)((ND + 255) / 256), 256, 0, stream>>>(bufA, embeds, bufB, ND);
  // 5) ordAcc = spmm(order)
  spmm_ord_kernel<<<(E + 255) / 256, 256, 0, stream>>>(rows, cols, order, ordAcc, E);
  // 6) acc2 = spmm(fst)
  {
    long long tt = (long long)E * 16;
    spmm_feat_kernel<<<(unsigned)((tt + 255) / 256), 256, 0, stream>>>(rows, cols, bufB, bufA, E);
  }
  // 7) scores via WMMA (one wave per 16-node tile, 8 waves/block)
  {
    int ntiles = (N + 15) / 16;
    int blocks = (ntiles + 7) / 8;
    score_wmma_kernel<<<blocks, 256, 0, stream>>>(bufA, order, ordAcc, embeds, scores, N, ntiles);
  }
  // 8) top-4096: radix select (3 levels) + compact + bitonic sort
  for (int lvl = 0; lvl < 3; ++lvl) {
    hist_kernel<<<1024, 256, 0, stream>>>(scores, hist, sel, lvl, N);
    scan_kernel<<<1, 1024, 0, stream>>>(hist, sel, lvl);
  }
  compact_kernel<<<(N + 255) / 256, 256, 0, stream>>>(scores, sel, candKey, candIdx, N);
  sort_kernel<<<1, 1024, 0, stream>>>(candKey, candIdx, outSeeds);
}
